// NonLocalBlock_12927851561069
// MI455X (gfx1250) — compile-verified, matrix operand order checked by reading, not run on hardware
//
#include <hip/hip_runtime.h>
#include <hip/hip_bf16.h>

// ---------------------------------------------------------------------------
// NonLocalBlock for MI455X (gfx1250, wave32, WMMA).
// B=4, C=64, IC=32, H=W=96 -> HW=9216, token N=4608, K(feature)=64.
// Two-pass column-softmax attention with bf16 WMMA (f32 accumulate).
// Compute-bound (~33 GFLOP on ~40 MB L2-resident data), so all operands are
// loaded directly into WMMA fragment registers (contiguous 16B per lane).
// ---------------------------------------------------------------------------

typedef __bf16 bf16_t;
typedef __attribute__((ext_vector_type(16))) __bf16 v16bf;
typedef __attribute__((ext_vector_type(8)))  __bf16 v8bf;
typedef __attribute__((ext_vector_type(8)))  float  v8f;

#define B_    4
#define C_    64
#define IC_   32
#define HW_   9216
#define N_    4608          // tokens per batch after [B,C,-1] reshape
#define NTILES_ (N_/16)     // 288

// ---------------------------------------------------------------------------
// Fragment gather helper.
// 16-bit A-matrix 16x32 layout (ISA 7.12.2): lane half h (lane>>4) holds
//   elements 0..7  -> K = 8h + {0..7}
//   elements 8..15 -> K = 16 + 8h + {0..7}
// B-matrix 32x16 mirrors this with N=lane%16.  When the source row (over K)
// is contiguous in memory, the fragment is exactly two 16B vector loads
// (the compiler fuses them into global_load_b128 / ds_load_b128).
// ---------------------------------------------------------------------------
__device__ __forceinline__ v16bf load_row_frag(const bf16_t* __restrict__ row, int half) {
    v8bf lo = *(const v8bf*)(row + (half << 3));
    v8bf hi = *(const v8bf*)(row + 16 + (half << 3));
    v16bf f;
#pragma unroll
    for (int i = 0; i < 8; ++i) { f[i] = lo[i]; f[i + 8] = hi[i]; }
    return f;
}

__device__ __forceinline__ v8f wmma_bf16(v16bf a, v16bf b, v8f c) {
    // D = A(16x32 bf16) * B(32x16 bf16) + C(16x16 f32)
    return __builtin_amdgcn_wmma_f32_16x16x32_bf16(
        /*neg_a=*/false, a, /*neg_b=*/false, b,
        /*c_mod=*/(short)0, c, /*reuse_a=*/false, /*reuse_b=*/false);
}

// Wave-private LDS ordering: same-wave LDS ops are processed in order by the
// LDS unit (DScnt, in-order), so a store->load RAW inside one wave needs no
// workgroup barrier -- only a compiler fence so IR/scheduler don't reorder.
__device__ __forceinline__ void wave_lds_fence() {
    __builtin_amdgcn_sched_barrier(0);
    asm volatile("" ::: "memory");
}

// ---------------------------------------------------------------------------
// Kernel 1: 1x1 conv (channel GEMM, K=64) in fp32, emit bf16.
//   proj[o][s] = sum_cc w[o][cc] * in[b][cc][s]      (o<32, s<9216)
// Reshape [32,9216]->[64,4608] is a memory no-op: c = 2o + (s>=4608), n = s%4608.
// transposed=1 -> out[b][n][64] rows (ThetaT/PhiT); 0 -> out[b][64][n] (G).
// ---------------------------------------------------------------------------
__global__ __launch_bounds__(256)
void proj_kernel(const float* __restrict__ in, const float* __restrict__ w,
                 bf16_t* __restrict__ out, int transposed) {
    const int idx = blockIdx.x * 256 + threadIdx.x;      // over B*64*4608
    const int n = idx % N_;
    const int c = (idx / N_) & 63;
    const int b = idx / (N_ * C_);
    const int o = c >> 1;
    const int s = (c & 1) * N_ + n;
    const float* xr = in + (size_t)b * C_ * HW_ + s;
    const float* wr = w + o * C_;
    float acc = 0.0f;
#pragma unroll 16
    for (int cc = 0; cc < C_; ++cc) acc += wr[cc] * xr[(size_t)cc * HW_];
    const bf16_t v = (bf16_t)acc;
    if (transposed) out[((size_t)b * N_ + n) * C_ + c] = v;
    else            out[((size_t)b * C_ + c) * N_ + n] = v;
}

// ---------------------------------------------------------------------------
// Kernel 2: column softmax stats.
// scores[n][m] = sum_c ThetaT[n][c] * PhiT[m][c]; reduce max / sum-exp over n
// for each column m.  Block = 8 waves, one (b, 16-wide m-tile) per block.
// C/D layout: lane L holds column m = L%16, rows n = v + 8*(L>>4), v=0..7.
// ---------------------------------------------------------------------------
__global__ __launch_bounds__(256)
void colstats_kernel(const bf16_t* __restrict__ thetaT, const bf16_t* __restrict__ phiT,
                     float* __restrict__ colmax, float* __restrict__ colsum) {
    const int b     = blockIdx.y;
    const int mtile = blockIdx.x;              // 0..287
    const int lane  = threadIdx.x & 31;
    const int wave  = threadIdx.x >> 5;        // 0..7
    const int half  = lane >> 4;
    const int r     = lane & 15;

    const bf16_t* thB = thetaT + (size_t)b * N_ * C_;
    const bf16_t* phB = phiT   + (size_t)b * N_ * C_;

    // B fragments (Phi, fixed m-tile): loop invariant.
    const bf16_t* phRow = phB + (size_t)(mtile * 16 + r) * C_;
    const v16bf pb0 = load_row_frag(phRow, half);        // c = 0..31
    const v16bf pb1 = load_row_frag(phRow + 32, half);   // c = 32..63

    float mx = -3.0e38f, sm = 0.0f;
    for (int nt = wave; nt < NTILES_; nt += 8) {
        const bf16_t* thRow = thB + (size_t)(nt * 16 + r) * C_;
        v16bf a0 = load_row_frag(thRow, half);
        v16bf a1 = load_row_frag(thRow + 32, half);
        v8f s = {0.f, 0.f, 0.f, 0.f, 0.f, 0.f, 0.f, 0.f};
        s = wmma_bf16(a0, pb0, s);
        s = wmma_bf16(a1, pb1, s);
        // online column stats over this lane's 8 rows
        float tm = s[0];
#pragma unroll
        for (int v = 1; v < 8; ++v) tm = fmaxf(tm, s[v]);
        const float nm = fmaxf(mx, tm);
        float add = 0.0f;
#pragma unroll
        for (int v = 0; v < 8; ++v) add += __expf(s[v] - nm);
        sm = sm * __expf(mx - nm) + add;
        mx = nm;
    }
    // combine the two half-wave row groups (same column m)
    const float mx2 = __shfl_xor(mx, 16, 32);
    const float sm2 = __shfl_xor(sm, 16, 32);
    const float M = fmaxf(mx, mx2);
    const float S = sm * __expf(mx - M) + sm2 * __expf(mx2 - M);

    __shared__ float smx[8][16];
    __shared__ float ssm[8][16];
    if (half == 0) { smx[wave][r] = M; ssm[wave][r] = S; }
    __syncthreads();
    if (threadIdx.x < 16) {
        float Mf = smx[0][threadIdx.x];
#pragma unroll
        for (int wv = 1; wv < 8; ++wv) Mf = fmaxf(Mf, smx[wv][threadIdx.x]);
        float Sf = 0.0f;
#pragma unroll
        for (int wv = 0; wv < 8; ++wv) Sf += ssm[wv][threadIdx.x] * __expf(smx[wv][threadIdx.x] - Mf);
        const int m = mtile * 16 + threadIdx.x;
        colmax[b * N_ + m] = Mf;
        colsum[b * N_ + m] = Sf;
    }
}

// ---------------------------------------------------------------------------
// Kernel 3: out^T[c][n] = sum_m P[n][m] * G[c][m],
//   P[n][m] = exp(scores[n][m] - colmax[m]) / colsum[m]   (scores recomputed).
// Block = 4 fully independent waves; each owns one 16-row n-tile, the full
// 64-c output block, and a private LDS staging slab.  m swept in steps of 32:
// two score tiles (2 WMMA) -> P staged in wave-private LDS -> one A-fragment
// -> 4 WMMAs against G B-fragments.  No workgroup barriers: same-wave LDS
// ordering is guaranteed by hardware (in-order DS pipeline).
// ---------------------------------------------------------------------------
__global__ __launch_bounds__(128)
void attn_out_kernel(const bf16_t* __restrict__ thetaT, const bf16_t* __restrict__ phiT,
                     const bf16_t* __restrict__ G,
                     const float* __restrict__ colmax, const float* __restrict__ colsum,
                     bf16_t* __restrict__ outT) {
    const int b     = blockIdx.y;
    const int wave  = threadIdx.x >> 5;                  // 0..3
    const int ntile = blockIdx.x * 4 + wave;             // 0..287
    const int lane  = threadIdx.x & 31;
    const int half  = lane >> 4;
    const int r     = lane & 15;

    __shared__ __align__(16) bf16_t Pst[4][16][40];      // 40-elem row stride (bank pad)

    const bf16_t* thB = thetaT + (size_t)b * N_ * C_;
    const bf16_t* phB = phiT   + (size_t)b * N_ * C_;
    const bf16_t* gB  = G      + (size_t)b * C_ * N_;
    const float*  cmB = colmax + b * N_;
    const float*  csB = colsum + b * N_;

    // Theta A-fragments for this n-tile: loop invariant.
    const bf16_t* thRow = thB + (size_t)(ntile * 16 + r) * C_;
    const v16bf ta0 = load_row_frag(thRow, half);
    const v16bf ta1 = load_row_frag(thRow + 32, half);

    v8f acc[4];
#pragma unroll
    for (int ct = 0; ct < 4; ++ct)
        acc[ct] = (v8f){0.f, 0.f, 0.f, 0.f, 0.f, 0.f, 0.f, 0.f};

    for (int m0 = 0; m0 < N_; m0 += 32) {
#pragma unroll
        for (int sub = 0; sub < 2; ++sub) {
            const int m = m0 + sub * 16 + r;             // this lane's column
            const bf16_t* phRow = phB + (size_t)m * C_;
            v16bf pb0 = load_row_frag(phRow, half);
            v16bf pb1 = load_row_frag(phRow + 32, half);
            v8f s = {0.f, 0.f, 0.f, 0.f, 0.f, 0.f, 0.f, 0.f};
            s = wmma_bf16(ta0, pb0, s);
            s = wmma_bf16(ta1, pb1, s);
            const float cm = cmB[m];
            const float rs = 1.0f / csB[m];
#pragma unroll
            for (int v = 0; v < 8; ++v) {
                const float p = __expf(s[v] - cm) * rs;
                Pst[wave][v + 8 * half][sub * 16 + r] = (bf16_t)p;
            }
        }
        wave_lds_fence();   // store->load RAW within this wave only
        // Reload P as an A-fragment (row n = r, K = local m 0..31): already
        // row-major in LDS, so two 16B ds loads -- no transpose needed.
        const v16bf pa = load_row_frag(&Pst[wave][r][0], half);
#pragma unroll
        for (int ct = 0; ct < 4; ++ct) {
            const bf16_t* gRow = gB + (size_t)(ct * 16 + r) * N_ + m0;  // G[c][m0..]
            v16bf gb = load_row_frag(gRow, half);
            acc[ct] = wmma_bf16(pa, gb, acc[ct]);
        }
        wave_lds_fence();   // load->store WAR before next iteration's stores
    }

    // D layout: lane column c = ct*16 + r, rows n = ntile*16 + v + 8*half.
#pragma unroll
    for (int ct = 0; ct < 4; ++ct) {
        const size_t base = ((size_t)b * C_ + (ct * 16 + r)) * N_ + ntile * 16 + 8 * half;
#pragma unroll
        for (int v = 0; v < 8; ++v)
            outT[base + v] = (bf16_t)acc[ct][v];
    }
}

// ---------------------------------------------------------------------------
// Kernel 4: mask = 1x1 conv (K=32) over out^T (viewed [B][32][9216]) + x.
// ---------------------------------------------------------------------------
__global__ __launch_bounds__(256)
void mask_kernel(const float* __restrict__ x, const float* __restrict__ w_mask,
                 const bf16_t* __restrict__ outT, float* __restrict__ out) {
    const int idx = blockIdx.x * 256 + threadIdx.x;      // over B*64*9216
    const int s = idx % HW_;
    const int o = (idx / HW_) & 63;
    const int b = idx / (HW_ * C_);
    const bf16_t* orow = outT + (size_t)b * IC_ * HW_ + s;
    const float* wr = w_mask + o * IC_;
    float acc = x[idx];
#pragma unroll 8
    for (int cc = 0; cc < IC_; ++cc) acc += wr[cc] * (float)orow[(size_t)cc * HW_];
    out[idx] = acc;
}

// ---------------------------------------------------------------------------
// Launch.  Inputs: x, y, w_phi, w_theta, w_g, w_mask (all fp32).
// Workspace layout (bytes):
//   PhiT   bf16 [B][N][64]   @ 0          (2,359,296)
//   ThetaT bf16 [B][N][64]   @ 2,359,296
//   G      bf16 [B][64][N]   @ 4,718,592
//   OutT   bf16 [B][64][N]   @ 7,077,888
//   colmax f32  [B][N]       @ 9,437,184
//   colsum f32  [B][N]       @ 9,510,912   (total ~9.6 MB)
// ---------------------------------------------------------------------------
extern "C" void kernel_launch(void* const* d_in, const int* in_sizes, int n_in,
                              void* d_out, int out_size, void* d_ws, size_t ws_size,
                              hipStream_t stream) {
    (void)in_sizes; (void)n_in; (void)out_size; (void)ws_size;
    const float* x       = (const float*)d_in[0];
    const float* y       = (const float*)d_in[1];
    const float* w_phi   = (const float*)d_in[2];
    const float* w_theta = (const float*)d_in[3];
    const float* w_g     = (const float*)d_in[4];
    const float* w_mask  = (const float*)d_in[5];
    float* out = (float*)d_out;

    char* ws = (char*)d_ws;
    bf16_t* PhiT   = (bf16_t*)(ws + 0);
    bf16_t* ThetaT = (bf16_t*)(ws + 2359296);
    bf16_t* G      = (bf16_t*)(ws + 4718592);
    bf16_t* OutT   = (bf16_t*)(ws + 7077888);
    float*  colmax = (float*)(ws + 9437184);
    float*  colsum = (float*)(ws + 9510912);

    proj_kernel<<<4608, 256, 0, stream>>>(x, w_phi,   PhiT,   1);
    proj_kernel<<<4608, 256, 0, stream>>>(y, w_theta, ThetaT, 1);
    proj_kernel<<<4608, 256, 0, stream>>>(x, w_g,     G,      0);
    colstats_kernel<<<dim3(NTILES_, B_), 256, 0, stream>>>(ThetaT, PhiT, colmax, colsum);
    attn_out_kernel<<<dim3(NTILES_ / 4, B_), 128, 0, stream>>>(ThetaT, PhiT, G, colmax, colsum, OutT);
    mask_kernel<<<9216, 256, 0, stream>>>(x, w_mask, OutT, out);
}
